// PointerDecoder_73572789780589
// MI455X (gfx1250) — matrix operand order repaired and checked by reading, not compiled, not run on hardware
//
#include <hip/hip_runtime.h>
#include <hip/hip_bf16.h>
#include <stdint.h>
#include <math.h>

// ---------------------------------------------------------------------------
// Problem constants: B=128, N=256 slots (+1 end), H=1024, T=257 steps
// ---------------------------------------------------------------------------
#define PB   128
#define PH   1024
#define PSLOT 257
#define PT   257

typedef __attribute__((ext_vector_type(16))) __bf16 v16bf;
typedef __attribute__((ext_vector_type(8)))  float  v8f;

#define DEV __device__ __forceinline__

DEV uint16_t f32_to_bf16_rne(float x) {
  uint32_t u = __float_as_uint(x);
  return (uint16_t)((u + 0x7FFFu + ((u >> 16) & 1u)) >> 16);
}
DEV void split_f32_bf16(float x, uint16_t &hi, uint16_t &lo) {
  uint32_t u  = __float_as_uint(x);
  uint32_t hr = (u + 0x7FFFu + ((u >> 16) & 1u)) & 0xFFFF0000u;
  hi = (uint16_t)(hr >> 16);
  lo = f32_to_bf16_rne(x - __uint_as_float(hr));
}

union Frag16 { v16bf v; uint32_t u[8]; uint4 q[2]; };

// CDNA5 async global->LDS copy (ASYNCcnt-tracked, bypasses VGPRs).
// VDST VGPR holds the 32-bit LDS byte address (= low 32 bits of the generic
// pointer to a __shared__ object, per the LDS aperture rule).
#define ASYNC_LDS_B128(ldsaddr, gptr)                                          \
  asm volatile("global_load_async_to_lds_b128 %0, %1, off"                     \
               :: "v"(ldsaddr), "v"(gptr) : "memory")
#define WAIT_ASYNCCNT0() asm volatile("s_wait_asynccnt 0x0" ::: "memory")

// ---------------------------------------------------------------------------
// bf16x3 WMMA GEMM:  C[M][ldc] = A[M][K] * B^T   (B stored row-major [N][K])
// block tile 128(M) x 64(N), 8 waves of 32x32 each, K step 32,
// double-buffered LDS; B tiles arrive via global_load_async_to_lds_b128.
// MODE 0: A rows from A0 with leading dim lda (f32, split to bf16 hi/lo here)
// MODE 1: "ext" rows: row r -> (b=r/257, s=r%257); s<256 -> encoder_outputs,
//         s==256 -> end_token (for the enc_Wa precompute)
// ---------------------------------------------------------------------------
template<int MODE>
__global__ __launch_bounds__(256) void gemm_bf16x3(
    const float* __restrict__ A0, const float* __restrict__ A1,
    const uint16_t* __restrict__ Bhi, const uint16_t* __restrict__ Blo,
    float* __restrict__ C, int K, int lda, int ldc)
{
  constexpr int KP = 40;  // padded LDS row (u16 units); keeps 16B alignment
  __shared__ uint16_t sAh[2][128*KP], sAl[2][128*KP];
  __shared__ uint16_t sBh[2][64*KP],  sBl[2][64*KP];

  const int tid  = threadIdx.x;
  const int bm   = blockIdx.x, bn = blockIdx.y;
  const int wave = tid >> 5,  lane = tid & 31;
  const int wm   = wave & 3,  wn   = wave >> 2;   // 4(M) x 2(N) wave grid
  const int mlane = lane & 15, half = lane >> 4;

  // per-thread B chunk: one 16B piece of hi and of lo per k-step
  const int br = tid >> 2;           // B row 0..63
  const int bc = (tid & 3) * 8;      // u16 col {0,8,16,24}

  auto stageB = [&](int k0, int buf) {
    size_t g = (size_t)(bn*64 + br)*(size_t)K + (size_t)(k0 + bc);
    uint32_t lh = (uint32_t)(uintptr_t)&sBh[buf][br*KP + bc];
    uint32_t ll = (uint32_t)(uintptr_t)&sBl[buf][br*KP + bc];
    ASYNC_LDS_B128(lh, Bhi + g);
    ASYNC_LDS_B128(ll, Blo + g);
  };

  auto stageA = [&](int k0, int buf) {
    #pragma unroll
    for (int i = 0; i < 16; ++i) {
      int idx = tid + i*256;              // 0..4095 = 128 rows x 32 k
      int r = idx >> 5, kk = idx & 31;
      int gr = bm*128 + r;
      float x;
      if (MODE == 0) {
        x = A0[(size_t)gr * (size_t)lda + (size_t)(k0 + kk)];
      } else {
        int bb = gr / PSLOT, ss = gr - bb*PSLOT;
        x = (ss < (PSLOT-1))
              ? A0[((size_t)(bb*(PSLOT-1) + ss))*PH + (size_t)(k0 + kk)]
              : A1[k0 + kk];
      }
      uint16_t h16, l16; split_f32_bf16(x, h16, l16);
      sAh[buf][r*KP + kk] = h16;
      sAl[buf][r*KP + kk] = l16;
    }
  };

  v8f acc[2][2] = {};

  stageB(0, 0);        // async copies in flight...
  stageA(0, 0);        // ...overlap with VALU split of the A tile
  WAIT_ASYNCCNT0();
  __syncthreads();

  int cur = 0;
  for (int k0 = 0; k0 < K; k0 += 32) {
    const int nxt = cur ^ 1;
    if (k0 + 32 < K) {      // prefetch next tile while computing this one
      stageB(k0 + 32, nxt);
      stageA(k0 + 32, nxt);
    }

    // ---- fragment loads (ISA 7.12.2 layouts) ----
    // A 16x32: lane m=lane&15; VGPR0-3 k=half*8+0..7, VGPR4-7 k=16+half*8+0..7
    Frag16 ah[2], al[2], bh[2], bl[2];
    #pragma unroll
    for (int tm = 0; tm < 2; ++tm) {
      int row = wm*32 + tm*16 + mlane;
      ah[tm].q[0] = *(const uint4*)&sAh[cur][row*KP + half*8];
      ah[tm].q[1] = *(const uint4*)&sAh[cur][row*KP + half*8 + 16];
      al[tm].q[0] = *(const uint4*)&sAl[cur][row*KP + half*8];
      al[tm].q[1] = *(const uint4*)&sAl[cur][row*KP + half*8 + 16];
    }
    // B 32x16: lane n=lane&15; VGPR0-3 k=half*8+0..7, VGPR4-7 k=16+half*8+0..7
    #pragma unroll
    for (int tn = 0; tn < 2; ++tn) {
      int row = wn*32 + tn*16 + mlane;
      bh[tn].q[0] = *(const uint4*)&sBh[cur][row*KP + half*8];
      bh[tn].q[1] = *(const uint4*)&sBh[cur][row*KP + half*8 + 16];
      bl[tn].q[0] = *(const uint4*)&sBl[cur][row*KP + half*8];
      bl[tn].q[1] = *(const uint4*)&sBl[cur][row*KP + half*8 + 16];
    }
    // ---- bf16x3: AhBh + AlBh + AhBl ----
    #pragma unroll
    for (int tm = 0; tm < 2; ++tm)
      #pragma unroll
      for (int tn = 0; tn < 2; ++tn) {
        acc[tm][tn] = __builtin_amdgcn_wmma_f32_16x16x32_bf16(
            false, ah[tm].v, false, bh[tn].v, (short)0, acc[tm][tn], false, false);
        acc[tm][tn] = __builtin_amdgcn_wmma_f32_16x16x32_bf16(
            false, al[tm].v, false, bh[tn].v, (short)0, acc[tm][tn], false, false);
        acc[tm][tn] = __builtin_amdgcn_wmma_f32_16x16x32_bf16(
            false, ah[tm].v, false, bl[tn].v, (short)0, acc[tm][tn], false, false);
      }

    WAIT_ASYNCCNT0();     // our async copies into nxt are done
    __syncthreads();      // everyone's copies/stores into nxt are done
    cur = nxt;
  }

  // ---- write out: C/D layout: lane n=lane&15, VGPR r -> m = half*8 + r ----
  // 32-bit indexing (max index ~33.7M) to avoid 64-bit address chains.
  const int base = (bm*128 + wm*32 + half*8)*ldc + (bn*64 + wn*32 + mlane);
  #pragma unroll
  for (int tm = 0; tm < 2; ++tm)
    #pragma unroll
    for (int tn = 0; tn < 2; ++tn) {
      float* p = C + (base + tm*16*ldc + tn*16);
      #pragma unroll
      for (int rr = 0; rr < 8; ++rr)
        p[rr*ldc] = acc[tm][tn][rr];
    }
}

// ---------------------------------------------------------------------------
// One-time weight packing: Wcat = [W_ih | W_hh]  -> bf16 hi/lo [4096][2048]
// ---------------------------------------------------------------------------
__global__ __launch_bounds__(256) void pack_wcat_kernel(
    const float* __restrict__ W_ih, const float* __restrict__ W_hh,
    uint16_t* __restrict__ hi, uint16_t* __restrict__ lo)
{
  int e = blockIdx.x*256 + threadIdx.x;          // < 4096*2048
  int n = e >> 11, k = e & 2047;
  float x = (k < PH) ? W_ih[(size_t)n*PH + k] : W_hh[(size_t)n*PH + (k - PH)];
  uint16_t h16, l16; split_f32_bf16(x, h16, l16);
  hi[e] = h16; lo[e] = l16;
}

__global__ __launch_bounds__(256) void split_kernel(
    const float* __restrict__ src, uint16_t* __restrict__ hi,
    uint16_t* __restrict__ lo, int count)
{
  int e = blockIdx.x*256 + threadIdx.x;
  if (e >= count) return;
  uint16_t h16, l16; split_f32_bf16(src[e], h16, l16);
  hi[e] = h16; lo[e] = l16;
}

// combined bias + state/mask init
__global__ __launch_bounds__(256) void init_kernel(
    const float* __restrict__ b_ih, const float* __restrict__ b_hh,
    const float* __restrict__ start_token,
    float* __restrict__ biasc, float* __restrict__ xh,
    float* __restrict__ cx, int* __restrict__ mask)
{
  int e = blockIdx.x*256 + threadIdx.x;          // < 128*2048
  int b = e >> 11, c = e & 2047;
  xh[e] = (c < PH) ? start_token[c] : 0.0f;      // x half = start token, h half = 0
  if (c >= PH) cx[b*PH + (c - PH)] = 0.0f;
  if (e < 4*PH) biasc[e] = b_ih[e] + b_hh[e];
  if (e < PB*PSLOT) mask[e] = 0;
}

// ---------------------------------------------------------------------------
// LSTM cell elementwise: gates [128][4096] (+bias), update cx, write hx into xh
// ---------------------------------------------------------------------------
__global__ __launch_bounds__(256) void lstm_cell_kernel(
    const float* __restrict__ gates, const float* __restrict__ bias,
    float* __restrict__ cx, float* __restrict__ xh)
{
  int e = blockIdx.x*256 + threadIdx.x;          // < 128*1024
  int b = e >> 10, h = e & 1023;
  const float* g = gates + (size_t)b*4*PH;
  float ig = g[h]        + bias[h];
  float fg = g[PH + h]   + bias[PH + h];
  float gg = g[2*PH + h] + bias[2*PH + h];
  float og = g[3*PH + h] + bias[3*PH + h];
  float si = 1.0f / (1.0f + expf(-ig));
  float sf = 1.0f / (1.0f + expf(-fg));
  float so = 1.0f / (1.0f + expf(-og));
  float c  = sf * cx[e] + si * tanhf(gg);
  cx[e] = c;
  xh[(size_t)b*2048 + PH + h] = so * tanhf(c);   // hx half of [x|h] buffer
}

// ---------------------------------------------------------------------------
// Attention scores: one wave per (b, slot); s = v . tanh(enc_Wa + uh)
// ---------------------------------------------------------------------------
__global__ __launch_bounds__(256) void scores_kernel(
    const float* __restrict__ encWa, const float* __restrict__ uh,
    const float* __restrict__ v, const int* __restrict__ mask,
    float* __restrict__ scores)
{
  int gw   = (blockIdx.x*256 + threadIdx.x) >> 5;
  int lane = threadIdx.x & 31;
  if (gw >= PB*PSLOT) return;
  int b = gw / PSLOT, s = gw - b*PSLOT;
  const float* e = encWa + ((size_t)b*PSLOT + s)*PH;
  const float* u = uh    + (size_t)b*PH;
  float acc = 0.0f;
  #pragma unroll
  for (int i = 0; i < 8; ++i) {
    int h = lane*4 + i*128;
    float4 ev = *(const float4*)(e + h);
    float4 uv = *(const float4*)(u + h);
    float4 vv = *(const float4*)(v + h);
    acc += vv.x * tanhf(ev.x + uv.x);
    acc += vv.y * tanhf(ev.y + uv.y);
    acc += vv.z * tanhf(ev.z + uv.z);
    acc += vv.w * tanhf(ev.w + uv.w);
  }
  #pragma unroll
  for (int off = 16; off > 0; off >>= 1) acc += __shfl_xor(acc, off, 32);
  if (lane == 0) scores[gw] = mask[gw] ? -INFINITY : acc;
}

// ---------------------------------------------------------------------------
// Per-batch: log_softmax -> d_out, argmax (first-max), mask update, gather x
// ---------------------------------------------------------------------------
__global__ __launch_bounds__(256) void finalize_kernel(
    const float* __restrict__ scores, const float* __restrict__ enc,
    const float* __restrict__ end_token, int* __restrict__ mask,
    float* __restrict__ xh, float* __restrict__ out_logp,
    int* __restrict__ out_idx, int t)
{
  int b = blockIdx.x, tid = threadIdx.x;
  __shared__ float sv[PSLOT];
  __shared__ float rv[256];
  __shared__ int   ri[256];
  __shared__ float rs[256];

  for (int s = tid; s < PSLOT; s += 256) sv[s] = scores[b*PSLOT + s];
  __syncthreads();

  float bv = -INFINITY; int bi = 0x7fffffff;
  for (int s = tid; s < PSLOT; s += 256) {
    float x = sv[s];
    if (x > bv || (x == bv && s < bi)) { bv = x; bi = s; }
  }
  rv[tid] = bv; ri[tid] = bi;
  __syncthreads();
  for (int off = 128; off > 0; off >>= 1) {
    if (tid < off) {
      float ov = rv[tid+off]; int oi = ri[tid+off];
      if (ov > rv[tid] || (ov == rv[tid] && oi < ri[tid])) { rv[tid] = ov; ri[tid] = oi; }
    }
    __syncthreads();
  }
  float mx = rv[0]; int idx = ri[0];

  float ps = 0.0f;
  for (int s = tid; s < PSLOT; s += 256) ps += expf(sv[s] - mx);
  rs[tid] = ps;
  __syncthreads();
  for (int off = 128; off > 0; off >>= 1) {
    if (tid < off) rs[tid] += rs[tid+off];
    __syncthreads();
  }
  float lse = mx + logf(rs[0]);

  for (int s = tid; s < PSLOT; s += 256)
    out_logp[((size_t)b*PT + t)*PSLOT + s] = sv[s] - lse;
  if (tid == 0) {
    out_idx[b*PT + t] = idx;
    if (idx < PSLOT-1) mask[b*PSLOT + idx] = 1;
  }
  const float* src = (idx < PSLOT-1) ? (enc + ((size_t)b*(PSLOT-1) + idx)*PH)
                                     : end_token;
  for (int h = tid; h < PH; h += 256) xh[(size_t)b*2048 + h] = src[h];
}

// ---------------------------------------------------------------------------
// Host orchestration
// ---------------------------------------------------------------------------
extern "C" void kernel_launch(void* const* d_in, const int* in_sizes, int n_in,
                              void* d_out, int out_size, void* d_ws, size_t ws_size,
                              hipStream_t stream) {
  (void)in_sizes; (void)n_in; (void)out_size; (void)ws_size;
  const float* enc   = (const float*)d_in[0];
  const float* W_ih  = (const float*)d_in[1];
  const float* W_hh  = (const float*)d_in[2];
  const float* b_ih  = (const float*)d_in[3];
  const float* b_hh  = (const float*)d_in[4];
  const float* Wa    = (const float*)d_in[5];
  const float* Ua    = (const float*)d_in[6];
  const float* v     = (const float*)d_in[7];
  const float* start = (const float*)d_in[8];
  const float* endt  = (const float*)d_in[9];

  char* ws = (char*)d_ws;
  size_t off = 0;
  auto take = [&](size_t bytes) { char* p = ws + off; off = (off + bytes + 255) & ~(size_t)255; return p; };
  uint16_t* wcat_hi = (uint16_t*)take((size_t)4096*2048*2);
  uint16_t* wcat_lo = (uint16_t*)take((size_t)4096*2048*2);
  uint16_t* ua_hi   = (uint16_t*)take((size_t)PH*PH*2);
  uint16_t* ua_lo   = (uint16_t*)take((size_t)PH*PH*2);
  uint16_t* wa_hi   = (uint16_t*)take((size_t)PH*PH*2);
  uint16_t* wa_lo   = (uint16_t*)take((size_t)PH*PH*2);
  float*    biasc   = (float*)take((size_t)4*PH*4);
  float*    encWa   = (float*)take((size_t)PB*PSLOT*PH*4);
  float*    gates   = (float*)take((size_t)PB*4*PH*4);
  float*    uh      = (float*)take((size_t)PB*PH*4);
  float*    xh      = (float*)take((size_t)PB*2048*4);
  float*    cx      = (float*)take((size_t)PB*PH*4);
  int*      mask    = (int*)take((size_t)PB*PSLOT*4);
  float*    scores  = (float*)take((size_t)PB*PSLOT*4);

  float* out_logp = (float*)d_out;
  int*   out_idx  = (int*)((float*)d_out + (size_t)PB*PT*PSLOT);

  // one-time: pack weights bf16 hi/lo, combined bias, state init
  pack_wcat_kernel<<<(4096*2048)/256, 256, 0, stream>>>(W_ih, W_hh, wcat_hi, wcat_lo);
  split_kernel<<<(PH*PH)/256, 256, 0, stream>>>(Ua, ua_hi, ua_lo, PH*PH);
  split_kernel<<<(PH*PH)/256, 256, 0, stream>>>(Wa, wa_hi, wa_lo, PH*PH);
  init_kernel<<<(PB*2048)/256, 256, 0, stream>>>(b_ih, b_hh, start, biasc, xh, cx, mask);

  // one-time: enc_Wa = ext @ Wa^T   (M = 128*257 = 32896 rows)
  gemm_bf16x3<1><<<dim3(PB*PSLOT/128, PH/64), 256, 0, stream>>>(
      enc, endt, wa_hi, wa_lo, encWa, PH, PH, PH);

  // sequential decode loop
  for (int t = 0; t < PT; ++t) {
    // gates = [x|hx] @ [W_ih|W_hh]^T   (M=128, N=4096, K=2048)
    gemm_bf16x3<0><<<dim3(1, 4*PH/64), 256, 0, stream>>>(
        xh, nullptr, wcat_hi, wcat_lo, gates, 2048, 2048, 4*PH);
    lstm_cell_kernel<<<(PB*PH)/256, 256, 0, stream>>>(gates, biasc, cx, xh);
    // uh = hx @ Ua^T   (A = hx half of xh buffer, lda=2048)
    gemm_bf16x3<0><<<dim3(1, PH/64), 256, 0, stream>>>(
        xh + PH, nullptr, ua_hi, ua_lo, uh, PH, 2048, PH);
    scores_kernel<<<(PB*PSLOT)/8, 256, 0, stream>>>(encWa, uh, v, mask, scores);
    finalize_kernel<<<PB, 256, 0, stream>>>(scores, enc, endt, mask, xh,
                                            out_logp, out_idx, t);
  }
}
